// TritonMLAMiniCPMAttention_83820581749277
// MI455X (gfx1250) — matrix operand order repaired and checked by reading, not compiled, not run on hardware
//
#include <hip/hip_runtime.h>
#include <hip/hip_bf16.h>
#include <math.h>

typedef __bf16 bf16;
typedef __attribute__((ext_vector_type(16))) __bf16 v16bf;
typedef __attribute__((ext_vector_type(8)))  __bf16 v8bf;
typedef __attribute__((ext_vector_type(8)))  float  v8f;

#define H_    16
#define S_    2048
#define QL_   1536
#define KVL_  512
#define RD_   64
#define VD_   128
#define HID_  2048
#define DQK_  576               /* KVL + RD */
#define SCALE_ 0.07216878364870323f   /* 1/sqrt(128+64) */
#define EPS_  1e-6f
#define NEG_  -1e30f

// ---------------------------------------------------------------------------
// Fragment helpers (layouts per CDNA5 ISA 7.12.2, wave32)
// ---------------------------------------------------------------------------
union FragU { v16bf v; v8bf h[2]; };

__device__ __forceinline__ v16bf load_frag(const bf16* p0, const bf16* p1) {
  FragU u;
  u.h[0] = *(const v8bf*)p0;
  u.h[1] = *(const v8bf*)p1;
  return u.v;
}

// A-fragment: two 8-elem chunks at off and off+16 (K and K+16 halves)
__device__ __forceinline__ v16bf load_fragA(const bf16* __restrict__ base, int off) {
  FragU u;
  u.h[0] = *(const v8bf*)(base + off);
  u.h[1] = *(const v8bf*)(base + off + 16);
  return u.v;
}

// B-fragment: two 8-elem chunks at off and off+16 (K and K+16 halves)
__device__ __forceinline__ v16bf load_fragB(const bf16* __restrict__ base, int off) {
  FragU u;
  u.h[0] = *(const v8bf*)(base + off);
  u.h[1] = *(const v8bf*)(base + off + 16);
  return u.v;
}

__device__ __forceinline__ v8f zero_acc() {
  v8f z;
#pragma unroll
  for (int i = 0; i < 8; ++i) z[i] = 0.0f;
  return z;
}

// ---------------------------------------------------------------------------
// Generic bf16 GEMM:  C[M,N] = A[M,K] * B[N,K]^T   (both row-major)
// Block: 128 threads = 4 waves (2x2). Block tile 128x128, wave tile 64x64.
// 16 WMMAs per 8 fragment loads per K-step. Uniform base pointers advance
// (scalar adds); per-lane offsets stay loop-invariant in VGPRs.
// FLAGS: bit0 = store bf16 (else f32), bit1 = transposed store C[n*ldc+m],
//        bit2 = N not multiple of 128 (clamp loads, guard stores)
// blockIdx.z = batch (strides sA/sB/sC in elements)
// Requires: M % 128 == 0, K % 32 == 0 (holds for all uses here).
// ---------------------------------------------------------------------------
template <int FLAGS>
__global__ __launch_bounds__(128) void gemm_bf16_kernel(
    const bf16* __restrict__ A, int lda, long long sA,
    const bf16* __restrict__ B, int ldb, long long sB,
    void* __restrict__ Cv, int ldc, long long sC,
    int M, int N, int K)
{
  const int lane = threadIdx.x & 31;
  const int wave = threadIdx.x >> 5;
  const long long bat = blockIdx.z;
  A += bat * sA;
  B += bat * sB;
  const int m0 = blockIdx.y * 128 + (wave >> 1) * 64;
  const int n0 = blockIdx.x * 128 + (wave & 1) * 64;
  const int hi = lane >> 4;
  const int lr = lane & 15;

  v8f acc[4][4];
#pragma unroll
  for (int i = 0; i < 4; ++i)
#pragma unroll
    for (int j = 0; j < 4; ++j) acc[i][j] = zero_acc();

  // Loop-invariant per-lane element offsets.
  int aoff[4], boff[4];
#pragma unroll
  for (int i = 0; i < 4; ++i)
    aoff[i] = (m0 + i * 16 + lr) * lda + hi * 8;
#pragma unroll
  for (int j = 0; j < 4; ++j) {
    int n = n0 + j * 16 + lr;
    if (FLAGS & 4) n = (n < N) ? n : (N - 1);  // clamp: garbage, never stored
    boff[j] = n * ldb + hi * 8;
  }

  // Uniform bases advance by 32 elements per K-step (scalar adds only).
  for (int kb = 0; kb < K; kb += 32, A += 32, B += 32) {
    v16bf a[4], b[4];
#pragma unroll
    for (int i = 0; i < 4; ++i) a[i] = load_fragA(A, aoff[i]);
#pragma unroll
    for (int j = 0; j < 4; ++j) b[j] = load_fragB(B, boff[j]);

    if (kb + 64 < K) {               // uniform; prefetch next-next tile
#pragma unroll
      for (int i = 0; i < 4; ++i) __builtin_prefetch(&A[aoff[i] + 64], 0, 1);
#pragma unroll
      for (int j = 0; j < 4; ++j) __builtin_prefetch(&B[boff[j] + 64], 0, 1);
    }

#pragma unroll
    for (int i = 0; i < 4; ++i)
#pragma unroll
      for (int j = 0; j < 4; ++j)
        acc[i][j] = __builtin_amdgcn_wmma_f32_16x16x32_bf16(
            false, a[i], false, b[j], (short)0, acc[i][j], false, false);
  }

#pragma unroll
  for (int i = 0; i < 4; ++i)
#pragma unroll
    for (int j = 0; j < 4; ++j)
#pragma unroll
      for (int e = 0; e < 8; ++e) {
        const int m = m0 + i * 16 + e + hi * 8;   // M multiple of 128
        const int n = n0 + j * 16 + lr;
        if (!(FLAGS & 4) || (n < N)) {
          const float v = acc[i][j][e];
          const size_t idx = (FLAGS & 2) ? ((size_t)n * ldc + m)
                                         : ((size_t)m * ldc + n);
          if (FLAGS & 1) ((bf16*)Cv)[bat * sC + idx] = (bf16)v;
          else           ((float*)Cv)[bat * sC + idx] = v;
        }
      }
}

// ---------------------------------------------------------------------------
// f32 -> bf16 convert
// ---------------------------------------------------------------------------
__global__ __launch_bounds__(256) void f32_to_bf16_kernel(
    const float* __restrict__ src, bf16* __restrict__ dst, int n)
{
  const int i = blockIdx.x * 256 + threadIdx.x;
  if (i < n) dst[i] = (bf16)src[i];
}

// ---------------------------------------------------------------------------
// Row RMSNorm: Y[row, 0:n] = bf16( X[row, 0:n] * rsqrt(mean(X^2)+eps) * g )
// ---------------------------------------------------------------------------
__global__ __launch_bounds__(256) void rmsnorm_kernel(
    const float* __restrict__ X, int ldx, int n,
    const float* __restrict__ g,
    bf16* __restrict__ Y, int ldy)
{
  __shared__ float red[256];
  const int row = blockIdx.x;
  const int tid = threadIdx.x;
  const float* x = X + (size_t)row * ldx;
  float s = 0.0f;
  for (int i = tid; i < n; i += 256) { const float v = x[i]; s += v * v; }
  red[tid] = s;
  __syncthreads();
  for (int w = 128; w > 0; w >>= 1) {
    if (tid < w) red[tid] += red[tid + w];
    __syncthreads();
  }
  const float inv = rsqrtf(red[0] / (float)n + EPS_);
  for (int i = tid; i < n; i += 256)
    Y[(size_t)row * ldy + i] = (bf16)(x[i] * inv * g[i]);
}

// ---------------------------------------------------------------------------
// k_rope: rope on kv_raw[t, 512:576] -> Kcat[t, 512:576]  (32 threads)
// ---------------------------------------------------------------------------
__global__ __launch_bounds__(32) void krope_kernel(
    const float* __restrict__ kv_raw, const int* __restrict__ pos_ids,
    bf16* __restrict__ Kcat)
{
  const int t = blockIdx.x;
  const int i = threadIdx.x;
  const float* x = kv_raw + (size_t)t * DQK_ + KVL_;
  const float x0 = x[i];
  const float x1 = x[i + 32];
  const float pos = (float)pos_ids[t];
  const float inv = expf(-(float)i * (logf(10000.0f) / 32.0f));
  const float c = cosf(pos * inv);
  const float sn = sinf(pos * inv);
  bf16* y = Kcat + (size_t)t * DQK_ + KVL_;
  y[i]      = (bf16)(x0 * c - x1 * sn);
  y[i + 32] = (bf16)(x1 * c + x0 * sn);
}

// ---------------------------------------------------------------------------
// Build Qcat[h][s][576] = [ qk_bf[s, h*512 : h*512+512] | rope(q_rope[s,h]) ]
// grid (S, H), 128 threads
// ---------------------------------------------------------------------------
__global__ __launch_bounds__(128) void build_qcat_kernel(
    const bf16* __restrict__ qk_bf,      // [S][H*KVL]
    const float* __restrict__ qrope_raw, // [S][H*RD]
    const int* __restrict__ pos_ids,
    bf16* __restrict__ Qcat)             // [H][S][DQK]
{
  const int s = blockIdx.x;
  const int h = blockIdx.y;
  const int tid = threadIdx.x;
  bf16* dst = Qcat + ((size_t)h * S_ + s) * DQK_;
  const bf16* src = qk_bf + (size_t)s * (H_ * KVL_) + h * KVL_;
#pragma unroll
  for (int k = tid; k < KVL_; k += 128) dst[k] = src[k];
  if (tid < 32) {
    const float* qr = qrope_raw + (size_t)s * (H_ * RD_) + h * RD_;
    const float x0 = qr[tid];
    const float x1 = qr[tid + 32];
    const float pos = (float)pos_ids[s];
    const float inv = expf(-(float)tid * (logf(10000.0f) / 32.0f));
    const float c = cosf(pos * inv);
    const float sn = sinf(pos * inv);
    dst[KVL_ + tid]      = (bf16)(x0 * c - x1 * sn);
    dst[KVL_ + 32 + tid] = (bf16)(x1 * c + x0 * sn);
  }
}

// ---------------------------------------------------------------------------
// Flash attention (one wave per 16-query block per head).
// scores = Qcat_h(16x576) @ Kcat^T in 32-key blocks, online softmax,
// O += P(16x32) @ Vt_h^T(32x128), final O/l -> Out[s][h*128+v] bf16.
// ---------------------------------------------------------------------------
__global__ __launch_bounds__(32) void mla_attention_kernel(
    const bf16* __restrict__ Qcat,  // [H][S][DQK]
    const bf16* __restrict__ Kcat,  // [S][DQK]
    const bf16* __restrict__ Vt,    // [H][VD][S]
    bf16* __restrict__ Out)         // [S][H*VD]
{
  const int h = blockIdx.y;
  const int qb = blockIdx.x * 16;
  const int lane = threadIdx.x & 31;
  const int hi = lane >> 4;
  const int lr = lane & 15;

  __shared__ __align__(16) bf16 P[16 * 32];

  // Q fragments held in registers: 18 x (16x32) bf16 covering K=576
  v16bf qf[18];
  {
    const int qoff = (h * S_ + qb + lr) * DQK_ + hi * 8;
#pragma unroll
    for (int d = 0; d < 18; ++d)
      qf[d] = load_fragA(Qcat, qoff + d * 32);
  }

  v8f o[8];
#pragma unroll
  for (int f = 0; f < 8; ++f) o[f] = zero_acc();
  float mrow[8], lrow[8];
#pragma unroll
  for (int e = 0; e < 8; ++e) { mrow[e] = NEG_; lrow[e] = 0.0f; }

  const int koff = lr * DQK_ + hi * 8;              // per-lane, invariant
  const int voff = lr * S_ + hi * 8;                // per-lane, invariant
  const bf16* Kp = Kcat;                             // advances 32 rows/step
  const bf16* Vp = Vt + (size_t)h * VD_ * S_;        // advances 32 cols/step

  for (int kb = 0; kb < qb + 16; kb += 32, Kp += 32 * DQK_, Vp += 32) {
    v8f s0 = zero_acc(), s1 = zero_acc();
#pragma unroll
    for (int d = 0; d < 18; ++d) {
      const v16bf b0 = load_fragB(Kp, koff + d * 32);
      const v16bf b1 = load_fragB(Kp, koff + d * 32 + 16 * DQK_);
      s0 = __builtin_amdgcn_wmma_f32_16x16x32_bf16(
          false, qf[d], false, b0, (short)0, s0, false, false);
      s1 = __builtin_amdgcn_wmma_f32_16x16x32_bf16(
          false, qf[d], false, b1, (short)0, s1, false, false);
    }

    // online softmax (row i of C lives in one 16-lane half, VGPR e -> row e+8*hi)
#pragma unroll
    for (int e = 0; e < 8; ++e) {
      const int row = qb + e + hi * 8;
      const int c0 = kb + lr;
      const int c1 = c0 + 16;
      float v0 = s0[e] * SCALE_;
      float v1 = s1[e] * SCALE_;
      if (c0 > row) v0 = NEG_;
      if (c1 > row) v1 = NEG_;
      float t = fmaxf(v0, v1);
      t = fmaxf(t, __shfl_xor(t, 1));
      t = fmaxf(t, __shfl_xor(t, 2));
      t = fmaxf(t, __shfl_xor(t, 4));
      t = fmaxf(t, __shfl_xor(t, 8));
      const float mnew = fmaxf(mrow[e], t);
      const float alpha = __expf(mrow[e] - mnew);
      const float p0 = __expf(v0 - mnew);
      const float p1 = __expf(v1 - mnew);
      float rs = p0 + p1;
      rs += __shfl_xor(rs, 1);
      rs += __shfl_xor(rs, 2);
      rs += __shfl_xor(rs, 4);
      rs += __shfl_xor(rs, 8);
      lrow[e] = lrow[e] * alpha + rs;
      mrow[e] = mnew;
#pragma unroll
      for (int f = 0; f < 8; ++f) o[f][e] *= alpha;
      const int r = e + hi * 8;
      P[r * 32 + lr]      = (bf16)p0;
      P[r * 32 + lr + 16] = (bf16)p1;
    }
    __syncthreads();
    v16bf pa;
    {
      const bf16* prow = &P[lr * 32 + hi * 8];
      pa = load_frag(prow, prow + 16);
    }
    __syncthreads();

    // O(16x128) += P(16x32) @ V_block(32x128); Vt is [VD][S] so rows contiguous
#pragma unroll
    for (int f = 0; f < 8; ++f) {
      const v16bf vb = load_fragB(Vp, voff + f * 16 * S_);
      o[f] = __builtin_amdgcn_wmma_f32_16x16x32_bf16(
          false, pa, false, vb, (short)0, o[f], false, false);
    }
  }

#pragma unroll
  for (int f = 0; f < 8; ++f)
#pragma unroll
    for (int e = 0; e < 8; ++e) {
      const int row = qb + e + hi * 8;
      const int col = f * 16 + lr;
      Out[(size_t)row * (H_ * VD_) + h * VD_ + col] = (bf16)(o[f][e] / lrow[e]);
    }
}

// ---------------------------------------------------------------------------
// Host launcher
// ---------------------------------------------------------------------------
template <int FLAGS>
static void launch_gemm(const bf16* A, int lda, long long sA,
                        const bf16* B, int ldb, long long sB,
                        void* C, int ldc, long long sC,
                        int M, int N, int K, int batch, hipStream_t stream)
{
  dim3 grid((N + 127) / 128, M / 128, batch);
  gemm_bf16_kernel<FLAGS><<<grid, 128, 0, stream>>>(A, lda, sA, B, ldb, sB,
                                                    C, ldc, sC, M, N, K);
}

extern "C" void kernel_launch(void* const* d_in, const int* in_sizes, int n_in,
                              void* d_out, int out_size, void* d_ws, size_t ws_size,
                              hipStream_t stream)
{
  (void)in_sizes; (void)n_in; (void)out_size; (void)ws_size;

  const float* hidden     = (const float*)d_in[0];
  const int*   pos        = (const int*)d_in[1];
  const float* w_q_a      = (const float*)d_in[2];
  const float* g_q_a      = (const float*)d_in[3];
  const float* w_q_b_rope = (const float*)d_in[4];
  const float* w_kv_a     = (const float*)d_in[5];
  const float* g_kv_a     = (const float*)d_in[6];
  const float* w_qk_merge = (const float*)d_in[7];
  const float* w_v_b      = (const float*)d_in[8];
  const float* w_o        = (const float*)d_in[9];

  char* p = (char*)d_ws;
  auto alloc = [&](size_t bytes) -> void* {
    void* r = (void*)p;
    p += (bytes + 255) & ~(size_t)255;
    return r;
  };

  bf16* hidden_bf = (bf16*)alloc((size_t)S_ * HID_ * 2);
  bf16* wqa_bf    = (bf16*)alloc((size_t)QL_ * HID_ * 2);
  bf16* wkva_bf   = (bf16*)alloc((size_t)DQK_ * HID_ * 2);
  bf16* wqbr_bf   = (bf16*)alloc((size_t)(H_ * RD_) * QL_ * 2);
  bf16* wqkm_bf   = (bf16*)alloc((size_t)(H_ * KVL_) * QL_ * 2);
  bf16* wvb_bf    = (bf16*)alloc((size_t)H_ * VD_ * KVL_ * 2);
  bf16* wo_bf     = (bf16*)alloc((size_t)HID_ * (H_ * VD_) * 2);
  float* qa_raw   = (float*)alloc((size_t)S_ * QL_ * 4);
  float* kv_raw   = (float*)alloc((size_t)S_ * DQK_ * 4);
  bf16* qa_bf     = (bf16*)alloc((size_t)S_ * QL_ * 2);
  bf16* Kcat      = (bf16*)alloc((size_t)S_ * DQK_ * 2);
  float* qrope_raw = (float*)alloc((size_t)S_ * (H_ * RD_) * 4);
  bf16* qk_bf     = (bf16*)alloc((size_t)S_ * (H_ * KVL_) * 2);
  bf16* Qcat      = (bf16*)alloc((size_t)H_ * S_ * DQK_ * 2);
  bf16* Vt        = (bf16*)alloc((size_t)H_ * VD_ * S_ * 2);
  bf16* attn_bf   = (bf16*)alloc((size_t)S_ * (H_ * VD_) * 2);

  auto cvt = [&](const float* src, bf16* dst, size_t n) {
    f32_to_bf16_kernel<<<(unsigned)((n + 255) / 256), 256, 0, stream>>>(src, dst, (int)n);
  };

  // 1) bf16 conversions
  cvt(hidden,     hidden_bf, (size_t)S_ * HID_);
  cvt(w_q_a,      wqa_bf,    (size_t)QL_ * HID_);
  cvt(w_kv_a,     wkva_bf,   (size_t)DQK_ * HID_);
  cvt(w_q_b_rope, wqbr_bf,   (size_t)(H_ * RD_) * QL_);
  cvt(w_qk_merge, wqkm_bf,   (size_t)(H_ * KVL_) * QL_);
  cvt(w_v_b,      wvb_bf,    (size_t)H_ * VD_ * KVL_);
  cvt(w_o,        wo_bf,     (size_t)HID_ * (H_ * VD_));

  // 2) projections from hidden (f32 out; N=576 needs guard)
  launch_gemm<0>(hidden_bf, HID_, 0, wqa_bf,  HID_, 0, qa_raw, QL_,  0,
                 S_, QL_,  HID_, 1, stream);
  launch_gemm<4>(hidden_bf, HID_, 0, wkva_bf, HID_, 0, kv_raw, DQK_, 0,
                 S_, DQK_, HID_, 1, stream);

  // 3) rmsnorms + k_rope -> qa_bf, Kcat
  rmsnorm_kernel<<<S_, 256, 0, stream>>>(qa_raw, QL_, QL_, g_q_a, qa_bf, QL_);
  rmsnorm_kernel<<<S_, 256, 0, stream>>>(kv_raw, DQK_, KVL_, g_kv_a, Kcat, DQK_);
  krope_kernel<<<S_, 32, 0, stream>>>(kv_raw, pos, Kcat);

  // 4) q_rope (f32) and qk_merge (bf16) projections from qa
  launch_gemm<0>(qa_bf, QL_, 0, wqbr_bf, QL_, 0, qrope_raw, H_ * RD_, 0,
                 S_, H_ * RD_, QL_, 1, stream);
  launch_gemm<1>(qa_bf, QL_, 0, wqkm_bf, QL_, 0, qk_bf, H_ * KVL_, 0,
                 S_, H_ * KVL_, QL_, 1, stream);

  // 5) assemble Qcat (qk | rope(q_rope)) per head
  build_qcat_kernel<<<dim3(S_, H_), 128, 0, stream>>>(qk_bf, qrope_raw, pos, Qcat);

  // 6) V_h = ckv @ v_b_h^T, bf16 transposed store [H][VD][S] (batched)
  launch_gemm<3>(Kcat, DQK_, 0, wvb_bf, KVL_, (long long)VD_ * KVL_,
                 Vt, S_, (long long)VD_ * S_, S_, VD_, KVL_, H_, stream);

  // 7) flash attention -> attn_bf [S][H*VD]
  mla_attention_kernel<<<dim3(S_ / 16, H_), 32, 0, stream>>>(Qcat, Kcat, Vt, attn_bf);

  // 8) output projection -> d_out (f32)
  launch_gemm<0>(attn_bf, H_ * VD_, 0, wo_bf, H_ * VD_, 0, (float*)d_out, HID_, 0,
                 S_, HID_, H_ * VD_, 1, stream);
}
